// SonicMoEAdapter_60911226192105
// MI455X (gfx1250) — matrix operand order repaired and checked by reading, not compiled, not run on hardware
//
#include <hip/hip_runtime.h>

#define D_    1024
#define F_    2048
#define E_    7
#define T_    2048
#define TOPK_ 2

typedef __bf16 bf16;
typedef __attribute__((ext_vector_type(16))) __bf16 v16bf;
typedef __attribute__((ext_vector_type(8)))  __bf16 v8bf;
typedef __attribute__((ext_vector_type(8)))  float  v8f;
typedef __attribute__((ext_vector_type(4)))  float  v4f;

#define WMMA_BF16(A, B, C) \
  __builtin_amdgcn_wmma_f32_16x16x32_bf16(false, (A), false, (B), (short)0, (C), false, false)

// fast sigmoid: one v_exp + one v_rcp
__device__ __forceinline__ float fast_sigmoid(float g) {
  return __builtin_amdgcn_rcpf(1.0f + __expf(-g));
}

// ---- A fragment: bf16 row-major, row = lane&15.
// Lanes 0-15: K=0..15; lanes 16-31: K=16..31. Two contiguous 16B chunks per lane.
__device__ __forceinline__ v16bf load_a_frag(const bf16* row_k0, int halfsel) {
  const v8bf lo = *(const v8bf*)(row_k0 + halfsel * 16);
  const v8bf hi = *(const v8bf*)(row_k0 + halfsel * 16 + 8);
  return __builtin_shufflevector(lo, hi, 0,1,2,3,4,5,6,7,8,9,10,11,12,13,14,15);
}

// ---- B fragment from pre-converted bf16 weights stored [N][K] row-major:
// lane's column = lane&15; lanes 0-15 hold K=0..15, lanes 16-31 K=16..31 of the chunk.
// 16 consecutive bf16 per lane = one aligned 32B access (2 x b128).
__device__ __forceinline__ v16bf load_b_frag(const bf16* p) {
  return *(const v16bf*)p;
}

// ===================== GEMM1 + SwiGLU + combine gate =====================
// act[t,f] = silu(x·Wg[f]) * (x·Wu[f]) * combine[t,expert]
// Wave tile: M=32 (two A frags), N=16; B (gate+up) reused across both A frags.
__global__ __launch_bounds__(256)
void gemm1_swiglu_kernel(const bf16* __restrict__ xb,       // [T, D] bf16
                         const bf16* __restrict__ w2f,      // [2F, D] bf16 (pre-converted)
                         const float* __restrict__ combine, // [T, 8] f32 (slot 7 == 1.0)
                         int expert,
                         bf16* __restrict__ act)            // [T, F] bf16
{
  const int NT = F_ / 16;
  const int wave = blockIdx.x * 8 + threadIdx.y;
  const int tM = wave / NT, tN = wave % NT;
  const int lane = threadIdx.x;
  const int half = lane >> 4, n = lane & 15;
  const int t0 = tM * 32, f0 = tN * 16;

  v8f accG0 = {}, accG1 = {}, accU0 = {}, accU1 = {};
  const bf16* arow0 = xb + (size_t)(t0 + n) * D_;
  const bf16* arow1 = arow0 + (size_t)16 * D_;
  const bf16* grow  = w2f + (size_t)(f0 + n) * D_ + half * 16;
  const bf16* urow  = grow + (size_t)F_ * D_;

  for (int k0 = 0; k0 < D_; k0 += 32) {
    v16bf a0 = load_a_frag(arow0 + k0, half);
    v16bf a1 = load_a_frag(arow1 + k0, half);
    v16bf bg = load_b_frag(grow + k0);
    v16bf bu = load_b_frag(urow + k0);
    accG0 = WMMA_BF16(a0, bg, accG0);
    accU0 = WMMA_BF16(a0, bu, accU0);
    accG1 = WMMA_BF16(a1, bg, accG1);
    accU1 = WMMA_BF16(a1, bu, accU1);
  }

#pragma unroll
  for (int i = 0; i < 8; ++i) {
    int t = t0 + half * 8 + i;
    float g = accG0[i], u = accU0[i];
    float s = g * fast_sigmoid(g) * u * combine[t * 8 + expert];
    act[(size_t)t * F_ + f0 + n] = (__bf16)s;

    t += 16;
    g = accG1[i]; u = accU1[i];
    s = g * fast_sigmoid(g) * u * combine[t * 8 + expert];
    act[(size_t)t * F_ + f0 + n] = (__bf16)s;
  }
}

// ===================== GEMM2 down-projection (accumulating) =====================
// y[t,d] (+)= sum_f act[t,f] * wproj[d,f]   (wproj [D,F] row-major -> K-contig per column)
// Wave tile: M=32, N=32 (2 A frags x 2 B frags -> 4 WMMAs per K-chunk).
__global__ __launch_bounds__(256)
void gemm2_down_kernel(const bf16* __restrict__ act,   // [T, F] bf16
                       const bf16* __restrict__ wproj, // [D, F] bf16 (pre-converted)
                       float* __restrict__ y,          // [T, D] f32
                       int accumulate)
{
  const int NT = D_ / 32;
  const int wave = blockIdx.x * 8 + threadIdx.y;
  const int tM = wave / NT, tN = wave % NT;
  const int lane = threadIdx.x;
  const int half = lane >> 4, n = lane & 15;
  const int t0 = tM * 32, d0 = tN * 32;

  v8f acc00 = {}, acc01 = {}, acc10 = {}, acc11 = {};
  const bf16* arow0 = act + (size_t)(t0 + n) * F_;
  const bf16* arow1 = arow0 + (size_t)16 * F_;
  const bf16* brow0 = wproj + (size_t)(d0 + n) * F_ + half * 16;
  const bf16* brow1 = brow0 + (size_t)16 * F_;

  for (int k0 = 0; k0 < F_; k0 += 32) {
    v16bf a0 = load_a_frag(arow0 + k0, half);
    v16bf a1 = load_a_frag(arow1 + k0, half);
    v16bf b0 = load_b_frag(brow0 + k0);
    v16bf b1 = load_b_frag(brow1 + k0);
    acc00 = WMMA_BF16(a0, b0, acc00);
    acc01 = WMMA_BF16(a0, b1, acc01);
    acc10 = WMMA_BF16(a1, b0, acc10);
    acc11 = WMMA_BF16(a1, b1, acc11);
  }

#pragma unroll
  for (int i = 0; i < 8; ++i) {
    const int t = t0 + half * 8 + i;
    float* yp0 = y + (size_t)t * D_ + d0 + n;
    float* yp1 = yp0 + (size_t)16 * D_;
    float v00 = acc00[i], v01 = acc01[i];
    float v10 = acc10[i], v11 = acc11[i];
    if (accumulate) {
      v00 += yp0[0];  v01 += yp0[16];
      v10 += yp1[0];  v11 += yp1[16];
    }
    yp0[0]  = v00;  yp0[16] = v01;
    yp1[0]  = v10;  yp1[16] = v11;
  }
}

// ===================== Router: logits -> softmax -> top-2 gates + aux stats =====================
__global__ void router_kernel(const float* __restrict__ x, const float* __restrict__ rw,
                              float* __restrict__ combine, float* __restrict__ stats)
{
  const int t = blockIdx.x * blockDim.x + threadIdx.x;
  if (t >= T_) return;

  float acc[E_];
#pragma unroll
  for (int e = 0; e < E_; ++e) acc[e] = 0.f;
  const float* xr = x + (size_t)t * D_;
  for (int d = 0; d < D_; ++d) {
    const float xv = xr[d];
#pragma unroll
    for (int e = 0; e < E_; ++e) acc[e] += xv * rw[e * D_ + d];
  }

  float m = acc[0];
#pragma unroll
  for (int e = 1; e < E_; ++e) m = fmaxf(m, acc[e]);
  float p[E_], s = 0.f;
#pragma unroll
  for (int e = 0; e < E_; ++e) { p[e] = __expf(acc[e] - m); s += p[e]; }
  const float inv = 1.f / s;
#pragma unroll
  for (int e = 0; e < E_; ++e) p[e] *= inv;

  int i0 = 0;
#pragma unroll
  for (int e = 1; e < E_; ++e) if (p[e] > p[i0]) i0 = e;
  int i1 = (i0 == 0) ? 1 : 0;
#pragma unroll
  for (int e = 0; e < E_; ++e) if (e != i0 && p[e] > p[i1]) i1 = e;

  const float denom = p[i0] + p[i1];
  const float g0 = p[i0] / denom, g1 = p[i1] / denom;
#pragma unroll
  for (int e = 0; e < E_; ++e)
    combine[t * 8 + e] = (e == i0) ? g0 : ((e == i1) ? g1 : 0.f);
  combine[t * 8 + 7] = 1.0f;   // shared-expert slot: gate == 1 (branch-free epilogue)

  atomicAdd(&stats[i0], 1.f);
  atomicAdd(&stats[i1], 1.f);
#pragma unroll
  for (int e = 0; e < E_; ++e) atomicAdd(&stats[8 + e], p[e]);
}

// Vectorized f32 -> bf16 conversion: 8 elements/thread (2 x float4 load, 1 x 16B store).
// n must be a multiple of 8 (true for all buffers here).
__global__ void cvt_bf16_vec_kernel(const float* __restrict__ src, bf16* __restrict__ dst, int n) {
  const int i = (blockIdx.x * blockDim.x + threadIdx.x) * 8;
  if (i < n) {
    const v4f a = *(const v4f*)(src + i);
    const v4f b = *(const v4f*)(src + i + 4);
    v8bf o;
#pragma unroll
    for (int j = 0; j < 4; ++j) { o[j] = (__bf16)a[j]; o[4 + j] = (__bf16)b[j]; }
    *(v8bf*)(dst + i) = o;
  }
}

__global__ void init_stats_kernel(float* stats) {
  if (threadIdx.x < 16) stats[threadIdx.x] = 0.f;
}

__global__ void aux_kernel(const float* __restrict__ stats, float* __restrict__ out_aux) {
  float s = 0.f;
  for (int e = 0; e < E_; ++e) {
    const float tpe = stats[e] / (float)(T_ * TOPK_);   // mean over tokens of one-hot sum, / K
    const float mp  = stats[8 + e] / (float)T_;         // mean prob
    s += tpe * mp;
  }
  *out_aux = (float)E_ * s * 0.01f;
}

extern "C" void kernel_launch(void* const* d_in, const int* in_sizes, int n_in,
                              void* d_out, int out_size, void* d_ws, size_t ws_size,
                              hipStream_t stream) {
  const float* x             = (const float*)d_in[0];
  const float* router_w      = (const float*)d_in[1];
  const float* cfc_w         = (const float*)d_in[2];
  const float* cproj_w       = (const float*)d_in[3];
  const float* shared_fc_w   = (const float*)d_in[4];
  const float* shared_proj_w = (const float*)d_in[5];
  float* y = (float*)d_out;

  // workspace (~24.1 MB):
  //   xb      [T*D]   bf16   4 MB
  //   act     [T*F]   bf16   8 MB
  //   wfc_b   [2F*D]  bf16   8 MB  (per-expert, reused)
  //   wproj_b [D*F]   bf16   4 MB  (per-expert, reused)
  //   combine [T*8]   f32    64 KB
  //   stats   [16]    f32
  char* wsp = (char*)d_ws;
  bf16*  xb      = (bf16*)wsp;
  bf16*  act     = (bf16*)(wsp + (size_t)T_ * D_ * 2);
  bf16*  wfc_b   = (bf16*)(wsp + (size_t)T_ * D_ * 2 + (size_t)T_ * F_ * 2);
  bf16*  wproj_b = wfc_b + (size_t)2 * F_ * D_;
  float* combine = (float*)(wproj_b + (size_t)D_ * F_);
  float* stats   = combine + (size_t)T_ * 8;

  const int NFC   = 2 * F_ * D_;   // fc weight elements per expert
  const int NPROJ = D_ * F_;       // proj weight elements per expert

  init_stats_kernel<<<1, 32, 0, stream>>>(stats);
  cvt_bf16_vec_kernel<<<(T_ * D_ / 8 + 255) / 256, 256, 0, stream>>>(x, xb, T_ * D_);
  router_kernel<<<(T_ + 63) / 64, 64, 0, stream>>>(x, router_w, combine, stats);

  const dim3 blk(32, 8);                                // 8 waves / block (wave32)
  const int g1_blocks = (T_ / 32) * (F_ / 16) / 8;      // 1024
  const int g2_blocks = (T_ / 32) * (D_ / 32) / 8;      // 256
  const int cfc_blocks   = (NFC / 8 + 255) / 256;
  const int cproj_blocks = (NPROJ / 8 + 255) / 256;

  for (int e = 0; e < E_; ++e) {
    cvt_bf16_vec_kernel<<<cfc_blocks, 256, 0, stream>>>(
        cfc_w + (size_t)e * NFC, wfc_b, NFC);
    cvt_bf16_vec_kernel<<<cproj_blocks, 256, 0, stream>>>(
        cproj_w + (size_t)e * NPROJ, wproj_b, NPROJ);
    gemm1_swiglu_kernel<<<g1_blocks, blk, 0, stream>>>(xb, wfc_b, combine, e, act);
    gemm2_down_kernel<<<g2_blocks, blk, 0, stream>>>(act, wproj_b, y, e > 0);
  }
  // shared expert: combine slot 7 holds 1.0
  cvt_bf16_vec_kernel<<<cfc_blocks, 256, 0, stream>>>(shared_fc_w, wfc_b, NFC);
  cvt_bf16_vec_kernel<<<cproj_blocks, 256, 0, stream>>>(shared_proj_w, wproj_b, NPROJ);
  gemm1_swiglu_kernel<<<g1_blocks, blk, 0, stream>>>(xb, wfc_b, combine, 7, act);
  gemm2_down_kernel<<<g2_blocks, blk, 0, stream>>>(act, wproj_b, y, 1);

  aux_kernel<<<1, 1, 0, stream>>>(stats, y + (size_t)T_ * D_);
}
